// Spatial_map_80951543595349
// MI455X (gfx1250) — matrix operand order repaired and checked by reading, not compiled, not run on hardware
//
#include <hip/hip_runtime.h>

#define C_DIM 512
#define F_DIM 64
#define HW_DIM 1024
#define BT_DIM 128

typedef __attribute__((ext_vector_type(16))) __bf16 bf16x16;
typedef __attribute__((ext_vector_type(16))) unsigned short u16x16;
typedef __attribute__((ext_vector_type(8))) float f32x8;
typedef __attribute__((ext_vector_type(4))) unsigned int u32x4;
typedef __attribute__((ext_vector_type(8))) int i32x8;
typedef __attribute__((ext_vector_type(4))) int i32x4;

__device__ __forceinline__ unsigned short f2bf(float f) {
  unsigned int u = __builtin_bit_cast(unsigned int, f);
  unsigned int r = u + 0x7FFFu + ((u >> 16) & 1u);  // round-to-nearest-even
  return (unsigned short)(r >> 16);
}

// ---------------------------------------------------------------------------
// TDM: DMA one 4KB A-fragment slab (4 mtiles x 512 bf16) global -> LDS.
// D# per ISA 8.3/8.4: group0 = {count=1, lds_addr, global_addr, type=2};
// group1: data_size=2B, tensor_dim0=8192, tensor_dim1=4, tile_dim0=512,
// tile_dim1=4, tensor_dim0_stride=8192 elements (mtile row pitch).
// ---------------------------------------------------------------------------
__device__ __forceinline__ void tdm_load_A(const unsigned short* gsrc, unsigned lds_off) {
  unsigned long long ga = (unsigned long long)(uintptr_t)gsrc;
  u32x4 g0;
  g0[0] = 1u;                                   // count=1 (valid user D#)
  g0[1] = lds_off;                              // lds_addr (bytes)
  g0[2] = (unsigned)ga;                         // global_addr[31:0]
  g0[3] = ((unsigned)(ga >> 32) & 0x01FFFFFFu)  // global_addr[56:32]
          | 0x80000000u;                        // type=2 ("image")
  i32x8 g1;
  g1[0] = (int)(1u << 16);                      // data_size=1 -> 2 bytes
  g1[1] = (int)(8192u << 16);                   // tensor_dim0[15:0] @ [63:48]
  g1[2] = (int)(4u << 16);                      // tensor_dim1[15:0] @ [95:80]
  g1[3] = (int)(512u << 16);                    // tile_dim0 @ [127:112]
  g1[4] = (int)4;                               // tile_dim1 @ [143:128]
  g1[5] = (int)8192;                            // tensor_dim0_stride[31:0]
  g1[6] = 0;
  g1[7] = 0;
  i32x4 gz = {};
#if __clang_major__ >= 23
  i32x8 gz8 = {};
  __builtin_amdgcn_tensor_load_to_lds(g0, g1, gz, gz, gz8, 0);
#else
  __builtin_amdgcn_tensor_load_to_lds(g0, g1, gz, gz, 0);
#endif
}

// ---------------------------------------------------------------------------
// k0: prep. wbar = mean_F(conv1_w) (folds the AvgPool into one C-vector),
// bbar = mean(conv1_b), and pack g3_w into the CDNA5 16x16x32 bf16 A-fragment
// layout so TDM can slab-copy fragments and lanes read contiguous 32B.
// A layout (contiguous k): lane<16 -> M=lane, K=j;
//                        lane>=16 -> same M, K shifted +16.
// ---------------------------------------------------------------------------
__global__ __launch_bounds__(512)
void k0_prep(const float* __restrict__ conv1_w, const float* __restrict__ conv1_b,
             const float* __restrict__ g3_w,
             float* __restrict__ wbar, unsigned short* __restrict__ packedA) {
  const int tid = threadIdx.x;
  if (tid < C_DIM) {
    float s = 0.f;
    for (int f = 0; f < F_DIM; ++f) s += conv1_w[f * C_DIM + tid];
    wbar[tid] = s * (1.0f / F_DIM);
  }
  if (tid == 0) {
    float s = 0.f;
    for (int f = 0; f < F_DIM; ++f) s += conv1_b[f];
    wbar[C_DIM] = s * (1.0f / F_DIM);
  }
  // packedA[mtile][ktile][lane][j], 4*16*32*16 bf16 = 64KB
  for (int i = 0; i < 64; ++i) {
    int idx = tid + 512 * i;
    int j = idx & 15;
    int lane = (idx >> 4) & 31;
    int ktile = (idx >> 9) & 15;
    int mtile = idx >> 13;
    int m = mtile * 16 + (lane & 15);
    int kk = j + ((lane >> 4) << 4);
    int c = ktile * 32 + kk;
    packedA[idx] = f2bf(g3_w[m * C_DIM + c]);
  }
}

// ---------------------------------------------------------------------------
// k1: streams x once. Per block: one (b,t) and a 128-wide p window.
// Stages bf16 x-tiles [p][k] in LDS (B fragments contiguous per lane),
// accumulates pooled = wbar.x during staging; wave 0 TDM-loads the A-fragment
// slab for the current k-chunk into LDS; bf16 WMMA for g3.
// ---------------------------------------------------------------------------
__global__ __launch_bounds__(256)
void k1_g3_pool(const float* __restrict__ x, const float* __restrict__ g3_b,
                const float* __restrict__ wbar, const unsigned short* __restrict__ packedA,
                float* __restrict__ g3_out, float* __restrict__ pooled_out) {
  __shared__ __align__(64) unsigned short ldsA[4 * 512];  // TDM dest: 4KB/chunk
  __shared__ __align__(64) unsigned short ldsB[128 * 32]; // [p][k] bf16, 8KB
  __shared__ float ldsW[C_DIM];
  __shared__ float ldsBias[F_DIM];
  __shared__ float ldsPool[128];

  const int tid = threadIdx.x;
  const int lane = tid & 31;
  const int wave = tid >> 5;        // 8 waves, each owns a 16-wide N (p) tile
  const int pbase = blockIdx.x * 128;
  const int bt = blockIdx.y;
  const int p_t = tid & 127;
  const int half = tid >> 7;        // 0: even k, 1: odd k

  for (int i = tid; i < C_DIM; i += 256) ldsW[i] = wbar[i];
  if (tid < F_DIM) ldsBias[tid] = g3_b[tid];
  const float bbar = wbar[C_DIM];
  const unsigned ldsA_off = (unsigned)(uintptr_t)(void*)&ldsA[0];

  const float* xbt = x + (size_t)bt * C_DIM * HW_DIM + pbase + p_t;

  f32x8 acc[4] = {};
  float poolAcc = 0.f;

  __syncthreads();

  for (int kc = 0; kc < 16; ++kc) {
    const int kbase = kc * 32;
    if (kc) __syncthreads();                       // protect LDS reuse
    if (wave == 0)                                 // DMA this chunk's A slab
      tdm_load_A(packedA + kc * 512, ldsA_off);
    if (kc + 1 < 16)
      __builtin_prefetch(xbt + (size_t)(kbase + 32) * HW_DIM, 0, 0);
#pragma unroll
    for (int e = 0; e < 16; ++e) {
      int k = 2 * e + half;
      float v = xbt[(size_t)(kbase + k) * HW_DIM]; // coalesced: lanes walk p
      poolAcc += ldsW[kbase + k] * v;              // fused pooled dot
      ldsB[p_t * 32 + k] = f2bf(v);
    }
    if (wave == 0)
      __builtin_amdgcn_s_wait_tensorcnt(0);        // A slab landed in LDS
    __syncthreads();
    // B fragment (32x16 K x N): lane<16 -> N=lane, K 0..15; lane>=16 -> K 16..31
    const int brow = wave * 16 + (lane & 15);
    const int koff = (lane >> 4) * 16;
    bf16x16 b = *(const bf16x16*)&ldsB[brow * 32 + koff];
#pragma unroll
    for (int m = 0; m < 4; ++m) {
      bf16x16 a = *(const bf16x16*)&ldsA[m * 512 + lane * 16];
      acc[m] = __builtin_amdgcn_wmma_f32_16x16x32_bf16(
          false, a, false, b, (short)0, acc[m], false, false);
    }
  }

  // pooled: combine even/odd-k halves, add bbar
  if (half == 1) ldsPool[p_t] = poolAcc;
  __syncthreads();
  if (half == 0)
    pooled_out[bt * HW_DIM + pbase + p_t] = poolAcc + ldsPool[p_t] + bbar;

  // g3 store (+bias). D layout: VGPR r -> M=r (+8 for hi lanes), N=lane%16
  const int ncol = pbase + wave * 16 + (lane & 15);
  const int fhi = (lane >> 4) << 3;
#pragma unroll
  for (int m = 0; m < 4; ++m) {
#pragma unroll
    for (int r = 0; r < 8; ++r) {
      int f = m * 16 + r + fhi;
      g3_out[((size_t)bt * F_DIM + f) * HW_DIM + ncol] = acc[m][r] + ldsBias[f];
    }
  }
}

// ---------------------------------------------------------------------------
// k2: z[bt][q] = sum_p pooled[bt][p] * ffnn_w[q][p], via bf16 WMMA.
// M=128(bt) -> 8 wave mtiles, N=1024(q) -> 64 blocks, K=1024 -> 32 steps.
// Both A (pooled rows) and B (ffnn_w rows) are K-contiguous in memory,
// matching the fragment layouts with direct contiguous per-lane loads.
// ---------------------------------------------------------------------------
__global__ __launch_bounds__(256)
void k2_zgemm(const float* __restrict__ pooled, const float* __restrict__ ffnn_w,
              float* __restrict__ zout) {
  const int tid = threadIdx.x;
  const int lane = tid & 31;
  const int mt = tid >> 5;                 // wave = bt tile
  const int hi = lane >> 4;
  const int btrow = mt * 16 + (lane & 15);
  const int q = blockIdx.x * 16 + (lane & 15);
  const float* prow = pooled + btrow * HW_DIM;
  const float* wrow = ffnn_w + (size_t)q * HW_DIM;

  f32x8 acc = {};
  for (int kc = 0; kc < 32; ++kc) {
    const int kbase = kc * 32;
    u16x16 au;
    const int ka = hi * 16;
#pragma unroll
    for (int j = 0; j < 8; ++j) au[j] = f2bf(prow[kbase + ka + j]);
#pragma unroll
    for (int j = 0; j < 8; ++j) au[8 + j] = f2bf(prow[kbase + 8 + ka + j]);
    bf16x16 a = __builtin_bit_cast(bf16x16, au);
    u16x16 bu;
    const int kb = hi * 16;
#pragma unroll
    for (int j = 0; j < 16; ++j) bu[j] = f2bf(wrow[kbase + kb + j]);
    bf16x16 b = __builtin_bit_cast(bf16x16, bu);
    acc = __builtin_amdgcn_wmma_f32_16x16x32_bf16(
        false, a, false, b, (short)0, acc, false, false);
  }
#pragma unroll
  for (int r = 0; r < 8; ++r) {
    int bt = mt * 16 + r + (hi << 3);
    zout[bt * HW_DIM + q] = acc[r];
  }
}

// ---------------------------------------------------------------------------
// k3: per-(b,t): z+bias -> relu/sigmoid/softmax blend s; out = s * g3.
// g3 (33.5MB) is L2-resident (192MB L2), so this pass is write-bandwidth bound.
// ---------------------------------------------------------------------------
__global__ __launch_bounds__(256)
void k3_act_scale(const float* __restrict__ z, const float* __restrict__ ffnn_b,
                  const float* __restrict__ actw, const float* __restrict__ g3,
                  float* __restrict__ out) {
  __shared__ float red[256];
  __shared__ float ldsS[HW_DIM];
  const int tid = threadIdx.x;
  const int bt = blockIdx.x;
  const float w0 = actw[0], w1 = actw[1], w2 = actw[2];

  float zv[4];
#pragma unroll
  for (int i = 0; i < 4; ++i) {
    int q = tid + 256 * i;
    zv[i] = z[bt * HW_DIM + q] + ffnn_b[q];
  }
  float mymax = fmaxf(fmaxf(zv[0], zv[1]), fmaxf(zv[2], zv[3]));
  red[tid] = mymax;
  __syncthreads();
  for (int s = 128; s > 0; s >>= 1) {
    if (tid < s) red[tid] = fmaxf(red[tid], red[tid + s]);
    __syncthreads();
  }
  const float zmax = red[0];
  __syncthreads();

  float ex[4], esum = 0.f;
#pragma unroll
  for (int i = 0; i < 4; ++i) { ex[i] = expf(zv[i] - zmax); esum += ex[i]; }
  red[tid] = esum;
  __syncthreads();
  for (int s = 128; s > 0; s >>= 1) {
    if (tid < s) red[tid] += red[tid + s];
    __syncthreads();
  }
  const float invS = 1.0f / red[0];
  __syncthreads();

#pragma unroll
  for (int i = 0; i < 4; ++i) {
    int q = tid + 256 * i;
    float zq = zv[i];
    ldsS[q] = w0 * fmaxf(zq, 0.f)
            + w1 * (1.0f / (1.0f + expf(-zq)))
            + w2 * (ex[i] * invS);
  }
  __syncthreads();

  const size_t base = (size_t)bt * F_DIM * HW_DIM;
  for (int it = 0; it < 256; ++it) {
    int flat = tid + 256 * it;
    out[base + flat] = ldsS[flat & (HW_DIM - 1)] * g3[base + flat];
  }
}

extern "C" void kernel_launch(void* const* d_in, const int* in_sizes, int n_in,
                              void* d_out, int out_size, void* d_ws, size_t ws_size,
                              hipStream_t stream) {
  (void)in_sizes; (void)n_in; (void)out_size; (void)ws_size;
  const float* x       = (const float*)d_in[0];
  const float* actw    = (const float*)d_in[2];
  const float* conv1_w = (const float*)d_in[3];
  const float* conv1_b = (const float*)d_in[4];
  const float* g3_w    = (const float*)d_in[5];
  const float* g3_b    = (const float*)d_in[6];
  const float* ffnn_w  = (const float*)d_in[7];
  const float* ffnn_b  = (const float*)d_in[8];
  float* out = (float*)d_out;

  // workspace layout (32B-aligned packedA)
  float* g3buf  = (float*)d_ws;                           // 8,388,608 f32
  float* pooled = g3buf + (size_t)BT_DIM * F_DIM * HW_DIM;// 131,072 f32
  float* zbuf   = pooled + BT_DIM * HW_DIM;               // 131,072 f32
  float* wbar   = zbuf + BT_DIM * HW_DIM;                 // 513 used, pad 520
  unsigned short* packedA = (unsigned short*)(wbar + 520);// 32,768 bf16

  k0_prep<<<1, 512, 0, stream>>>(conv1_w, conv1_b, g3_w, wbar, packedA);
  k1_g3_pool<<<dim3(8, 128), 256, 0, stream>>>(x, g3_b, wbar, packedA, g3buf, pooled);
  k2_zgemm<<<64, 256, 0, stream>>>(pooled, ffnn_w, zbuf);
  k3_act_scale<<<128, 256, 0, stream>>>(zbuf, ffnn_b, actw, g3buf, out);
}
